// Extractor_25640954757659
// MI455X (gfx1250) — compile-verified, matrix operand order checked by reading, not run on hardware
//
#include <hip/hip_runtime.h>
#include <hip/hip_bf16.h>

// ---------------- problem constants (from reference) ----------------
#define BB     8
#define HH_    32
#define WW_    32
#define DIM_   768
#define NH_    6
#define NP_    4
#define DV_    384
#define DH_    64
#define HID_   192
#define NTOK   5376          // 21 * (32*32/4)
#define LIN_   1024          // 32*32
#define MQ     (BB * NTOK)   // 43008 query rows
#define MF     (BB * LIN_)   // 8192 feat rows

typedef __attribute__((ext_vector_type(16))) __bf16 v16bf;
typedef __attribute__((ext_vector_type(8)))  __bf16 v8bf;
typedef __attribute__((ext_vector_type(8)))  float  v8f;

// ---------------- weight transpose + fp32 -> bf16 ----------------
// W is K x N row-major (as in reference); Wt is N x K row-major.
__global__ void wt_convert_kernel(const float* __restrict__ W,
                                  __bf16* __restrict__ Wt, int K, int N) {
    int i = blockIdx.x * blockDim.x + threadIdx.x;
    if (i >= K * N) return;
    int k = i % K;
    int n = i / K;
    Wt[(size_t)n * K + k] = (__bf16)W[(size_t)k * N + n];
}

// ---------------- LayerNorm -> bf16 (one wave32 per row) ----------------
__global__ void ln_bf16_kernel(const float* __restrict__ x,
                               const float* __restrict__ g,
                               const float* __restrict__ b,
                               __bf16* __restrict__ out, int rows, int cols) {
    int row = blockIdx.x * blockDim.y + threadIdx.y;
    if (row >= rows) return;
    int lane = threadIdx.x;
    const float* xr = x + (size_t)row * cols;
    float s = 0.f, s2 = 0.f;
    for (int c = lane; c < cols; c += 32) {
        float v = xr[c];
        s += v; s2 += v * v;
    }
    #pragma unroll
    for (int off = 16; off; off >>= 1) {
        s  += __shfl_xor(s,  off, 32);
        s2 += __shfl_xor(s2, off, 32);
    }
    float mu  = s / (float)cols;
    float var = s2 / (float)cols - mu * mu;
    float inv = rsqrtf(var + 1e-6f);
    __bf16* o = out + (size_t)row * cols;
    for (int c = lane; c < cols; c += 32)
        o[c] = (__bf16)((xr[c] - mu) * inv * g[c] + b[c]);
}

// load one 16x32 bf16 WMMA operand fragment (two contiguous 16B vectors)
__device__ __forceinline__ v16bf load_frag(const __bf16* p) {
    v8bf lo = *(const v8bf*)p;
    v8bf hi = *(const v8bf*)(p + 16);
    v16bf r;
    #pragma unroll
    for (int i = 0; i < 8; ++i) { r[i] = lo[i]; r[i + 8] = hi[i]; }
    return r;
}

// ---------------- bf16 WMMA GEMM: C(MxN) = A(MxK) * Bt(NxK)^T + bias (+res) ----
// One wave computes a 16x(16*NT) strip. The A fragment is loaded once per
// K-step and reused for NT B fragments / accumulators. Fragments are
// double-buffered so next-K loads overlap the current WMMAs (partial
// s_wait_loadcnt instead of a full drain). Block = 4 waves stacked along M.
template<int NT, bool RES>
__global__ void gemm_bf16_kernel(const __bf16* __restrict__ A,
                                 const __bf16* __restrict__ Bt,
                                 const float*  __restrict__ bias,
                                 const float*  __restrict__ res,
                                 float* __restrict__ C,
                                 int M, int N, int K) {
    int lane  = threadIdx.x;                 // 0..31
    int mtile = blockIdx.y * blockDim.y + threadIdx.y;
    if (mtile * 16 >= M) return;             // wave-uniform

    int l15   = lane & 15;
    int khalf = lane >> 4;                   // 0: K{0-7,16-23}, 1: K{8-15,24-31}
    int m     = mtile * 16 + l15;
    const __bf16* arow = A + (size_t)m * K + khalf * 8;

    int ncol[NT];
    const __bf16* brow[NT];
    v8f acc[NT];
    #pragma unroll
    for (int t = 0; t < NT; ++t) {
        ncol[t] = (blockIdx.x * NT + t) * 16 + l15;
        int nld = (ncol[t] < N) ? ncol[t] : (N - 1);   // clamp: EXEC stays all-1
        brow[t] = Bt + (size_t)nld * K + khalf * 8;
        acc[t]  = (v8f){};
    }

    // ---- software pipeline: prologue ----
    v16bf av = load_frag(arow);
    v16bf bv[NT];
    #pragma unroll
    for (int t = 0; t < NT; ++t) bv[t] = load_frag(brow[t]);

    // ---- steady state: issue k0+32 loads, then WMMA on k0 fragments ----
    for (int k0 = 0; k0 < K - 32; k0 += 32) {
        v16bf av_n = load_frag(arow + k0 + 32);
        v16bf bv_n[NT];
        #pragma unroll
        for (int t = 0; t < NT; ++t) bv_n[t] = load_frag(brow[t] + k0 + 32);
        #pragma unroll
        for (int t = 0; t < NT; ++t)
            acc[t] = __builtin_amdgcn_wmma_f32_16x16x32_bf16(
                         false, av, false, bv[t], (short)0, acc[t], false, false);
        av = av_n;
        #pragma unroll
        for (int t = 0; t < NT; ++t) bv[t] = bv_n[t];
    }
    // ---- epilogue: last K-step ----
    #pragma unroll
    for (int t = 0; t < NT; ++t)
        acc[t] = __builtin_amdgcn_wmma_f32_16x16x32_bf16(
                     false, av, false, bv[t], (short)0, acc[t], false, false);

    #pragma unroll
    for (int t = 0; t < NT; ++t) {
        if (ncol[t] < N) {
            float bs = bias[ncol[t]];
            #pragma unroll
            for (int r = 0; r < 8; ++r) {
                int row = mtile * 16 + r + khalf * 8;
                float v = acc[t][r] + bs;
                if (RES) v += res[(size_t)row * N + ncol[t]];
                C[(size_t)row * N + ncol[t]] = v;
            }
        }
    }
}

// ---------------- deformable sampling: one wave per (b, token, head) --------
__global__ void msda_sample_kernel(const float* __restrict__ v,      // (B,LIN,NH,DH)
                                   const float* __restrict__ off,    // (B,LQ,NH,NP,2)
                                   const float* __restrict__ lgt,    // (B,LQ,NH,NP)
                                   const float* __restrict__ ref,    // (B,LQ,2)
                                   __bf16* __restrict__ out) {       // (B,LQ,NH,DH) bf16
    int gw = blockIdx.x * blockDim.y + threadIdx.y;
    int total = BB * NTOK * NH_;
    if (gw >= total) return;
    int head = gw % NH_;
    int tq   = (gw / NH_) % NTOK;
    int b    = gw / (NH_ * NTOK);
    int lane = threadIdx.x;

    size_t qbase = (size_t)b * NTOK + tq;
    const float* lg = lgt + (qbase * NH_ + head) * NP_;
    float l0 = lg[0], l1 = lg[1], l2 = lg[2], l3 = lg[3];
    float mx = fmaxf(fmaxf(l0, l1), fmaxf(l2, l3));
    float e[NP_];
    e[0] = __expf(l0 - mx); e[1] = __expf(l1 - mx);
    e[2] = __expf(l2 - mx); e[3] = __expf(l3 - mx);
    float inv = 1.f / (e[0] + e[1] + e[2] + e[3]);

    const float* of = off + (qbase * NH_ + head) * NP_ * 2;
    float rx = ref[qbase * 2 + 0];
    float ry = ref[qbase * 2 + 1];

    float acc0 = 0.f, acc1 = 0.f;
    #pragma unroll
    for (int p = 0; p < NP_; ++p) {
        float aw = e[p] * inv;
        float X = (rx + of[2 * p + 0] * (1.f / WW_)) * WW_ - 0.5f;
        float Y = (ry + of[2 * p + 1] * (1.f / HH_)) * HH_ - 0.5f;
        float x0f = floorf(X), y0f = floorf(Y);
        float wx = X - x0f, wy = Y - y0f;
        int x0 = (int)x0f, y0 = (int)y0f;
        #pragma unroll
        for (int dy = 0; dy < 2; ++dy) {
            #pragma unroll
            for (int dx = 0; dx < 2; ++dx) {
                int xi = x0 + dx, yi = y0 + dy;
                if (xi >= 0 && xi < WW_ && yi >= 0 && yi < HH_) {
                    float wgt = (dx ? wx : 1.f - wx) * (dy ? wy : 1.f - wy) * aw;
                    const float* src = v +
                        (((size_t)b * LIN_ + (size_t)(yi * WW_ + xi)) * NH_ + head) * DH_;
                    acc0 += wgt * src[lane];
                    acc1 += wgt * src[lane + 32];
                }
            }
        }
    }
    __bf16* o = out + (qbase * NH_ + head) * DH_;
    o[lane]      = (__bf16)acc0;
    o[lane + 32] = (__bf16)acc1;
}

// ---------------- depthwise 3x3 (3-level pyramid) + exact GELU -> bf16 ------
__global__ void dwconv_gelu_kernel(const float* __restrict__ y,    // (B,LQ,HID)
                                   const float* __restrict__ wdw,  // (3,3,1,HID)
                                   const float* __restrict__ bdw,  // (HID)
                                   __bf16* __restrict__ out) {     // (B,LQ,HID) bf16
    int idx = blockIdx.x * blockDim.x + threadIdx.x;
    int total = BB * NTOK * HID_;
    if (idx >= total) return;
    int c = idx % HID_;
    int t = (idx / HID_) % NTOK;
    int b = idx / (HID_ * NTOK);

    int base, hh, ww;
    if (t < 4096)      { base = 0;    hh = 2 * HH_;  ww = 2 * WW_; }
    else if (t < 5120) { base = 4096; hh = HH_;      ww = WW_;     }
    else               { base = 5120; hh = HH_ / 2;  ww = WW_ / 2; }
    int loc = t - base;
    int yy = loc / ww, xx = loc % ww;

    float acc = bdw[c];
    #pragma unroll
    for (int dy = -1; dy <= 1; ++dy) {
        #pragma unroll
        for (int dx = -1; dx <= 1; ++dx) {
            int Y = yy + dy, X = xx + dx;
            if (Y >= 0 && Y < hh && X >= 0 && X < ww) {
                float wv = wdw[((dy + 1) * 3 + (dx + 1)) * HID_ + c];
                acc += wv * y[((size_t)b * NTOK + base + Y * ww + X) * HID_ + c];
            }
        }
    }
    float g = 0.5f * acc * (1.f + erff(acc * 0.70710678118654752f));
    out[idx] = (__bf16)g;
}

// =====================================================================
extern "C" void kernel_launch(void* const* d_in, const int* in_sizes, int n_in,
                              void* d_out, int out_size, void* d_ws, size_t ws_size,
                              hipStream_t stream) {
    const float* query = (const float*)d_in[0];
    const float* refp  = (const float*)d_in[1];
    const float* feat  = (const float*)d_in[2];
    const float* qn_g  = (const float*)d_in[5];
    const float* qn_b  = (const float*)d_in[6];
    const float* fn_g  = (const float*)d_in[7];
    const float* fn_b  = (const float*)d_in[8];
    const float* W_off = (const float*)d_in[9];
    const float* b_off = (const float*)d_in[10];
    const float* W_att = (const float*)d_in[11];
    const float* b_att = (const float*)d_in[12];
    const float* W_val = (const float*)d_in[13];
    const float* b_val = (const float*)d_in[14];
    const float* W_out = (const float*)d_in[15];
    const float* b_out = (const float*)d_in[16];
    const float* ffn_g = (const float*)d_in[17];
    const float* ffn_b = (const float*)d_in[18];
    const float* fc1_w = (const float*)d_in[19];
    const float* fc1_b = (const float*)d_in[20];
    const float* dw_w  = (const float*)d_in[21];
    const float* dw_b  = (const float*)d_in[22];
    const float* fc2_w = (const float*)d_in[23];
    const float* fc2_b = (const float*)d_in[24];
    float* outp = (float*)d_out;

    // ---- workspace layout (bytes, 256-aligned) ----
    char* ws = (char*)d_ws;
    size_t o = 0;
    auto take = [&](size_t bytes) { size_t p = o; o = (o + bytes + 255) & ~(size_t)255; return p; };
    __bf16* qln  = (__bf16*)(ws + take((size_t)MQ * DIM_ * 2));  // later reused as xln
    __bf16* fln  = (__bf16*)(ws + take((size_t)MF * DIM_ * 2));
    float*  vbuf = (float*) (ws + take((size_t)MF * DV_  * 4));
    float*  offb = (float*) (ws + take((size_t)MQ * NH_ * NP_ * 2 * 4));
    float*  awb  = (float*) (ws + take((size_t)MQ * NH_ * NP_ * 4));
    __bf16* samp = (__bf16*)(ws + take((size_t)MQ * DV_  * 2));
    float*  xbuf = (float*) (ws + take((size_t)MQ * DIM_ * 4));
    float*  ybuf = (float*) (ws + take((size_t)MQ * HID_ * 4));
    __bf16* ybf  = (__bf16*)(ws + take((size_t)MQ * HID_ * 2));
    __bf16* woffT = (__bf16*)(ws + take((size_t)DIM_ * NH_ * NP_ * 2 * 2));
    __bf16* wattT = (__bf16*)(ws + take((size_t)DIM_ * NH_ * NP_ * 2));
    __bf16* wvalT = (__bf16*)(ws + take((size_t)DIM_ * DV_ * 2));
    __bf16* woutT = (__bf16*)(ws + take((size_t)DV_  * DIM_ * 2));
    __bf16* wfc1T = (__bf16*)(ws + take((size_t)DIM_ * HID_ * 2));
    __bf16* wfc2T = (__bf16*)(ws + take((size_t)HID_ * DIM_ * 2));
    __bf16* xln = qln;  // qln dead after offset/attn GEMMs

    auto cvt = [&](const float* W, __bf16* Wt, int K, int N) {
        int n = K * N;
        wt_convert_kernel<<<(n + 255) / 256, 256, 0, stream>>>(W, Wt, K, N);
    };
    cvt(W_off, woffT, DIM_, NH_ * NP_ * 2);
    cvt(W_att, wattT, DIM_, NH_ * NP_);
    cvt(W_val, wvalT, DIM_, DV_);
    cvt(W_out, woutT, DV_,  DIM_);
    cvt(fc1_w, wfc1T, DIM_, HID_);
    cvt(fc2_w, wfc2T, HID_, DIM_);

    dim3 lnblk(32, 8);
    ln_bf16_kernel<<<MQ / 8, lnblk, 0, stream>>>(query, qn_g, qn_b, qln, MQ, DIM_);
    ln_bf16_kernel<<<MF / 8, lnblk, 0, stream>>>(feat,  fn_g, fn_b, fln, MF, DIM_);

    dim3 gblk(32, 4);
    auto gemm = [&](const __bf16* A, const __bf16* Bt, const float* bias,
                    const float* res, float* C, int M, int N, int K) {
        if (N >= 64) {                                   // NT=4 strip (N mult of 16)
            dim3 grid((N + 63) / 64, (M + 63) / 64);
            if (res)
                gemm_bf16_kernel<4, true ><<<grid, gblk, 0, stream>>>(A, Bt, bias, res, C, M, N, K);
            else
                gemm_bf16_kernel<4, false><<<grid, gblk, 0, stream>>>(A, Bt, bias, res, C, M, N, K);
        } else {                                         // narrow GEMMs (N=24/48)
            dim3 grid((N + 31) / 32, (M + 63) / 64);
            if (res)
                gemm_bf16_kernel<2, true ><<<grid, gblk, 0, stream>>>(A, Bt, bias, res, C, M, N, K);
            else
                gemm_bf16_kernel<2, false><<<grid, gblk, 0, stream>>>(A, Bt, bias, res, C, M, N, K);
        }
    };

    gemm(fln, wvalT, b_val, nullptr, vbuf, MF, DV_, DIM_);               // value proj
    gemm(qln, woffT, b_off, nullptr, offb, MQ, NH_ * NP_ * 2, DIM_);     // offsets
    gemm(qln, wattT, b_att, nullptr, awb,  MQ, NH_ * NP_, DIM_);         // attn logits

    {   // sampling: one wave per (b, token, head)
        int waves = BB * NTOK * NH_;
        dim3 blk(32, 8);
        msda_sample_kernel<<<(waves + 7) / 8, blk, 0, stream>>>(vbuf, offb, awb, refp, samp);
    }

    gemm(samp, woutT, b_out, query, xbuf, MQ, DIM_, DV_);                // out proj + residual

    ln_bf16_kernel<<<MQ / 8, lnblk, 0, stream>>>(xbuf, ffn_g, ffn_b, xln, MQ, DIM_);
    gemm(xln, wfc1T, fc1_b, nullptr, ybuf, MQ, HID_, DIM_);              // fc1

    {   // depthwise conv + gelu
        int n = BB * NTOK * HID_;
        dwconv_gelu_kernel<<<(n + 255) / 256, 256, 0, stream>>>(ybuf, dw_w, dw_b, ybf);
    }

    gemm(ybf, wfc2T, fc2_b, xbuf, outp, MQ, DIM_, HID_);                 // fc2 + residual
}
